// PositionWiseFeedForward_8272107012564
// MI455X (gfx1250) — matrix operand order repaired and checked by reading, not compiled, not run on hardware
//
#include <hip/hip_runtime.h>

// ---------------------------------------------------------------------------
// PositionWiseFeedForward (quantized FFN) for MI455X / gfx1250, wave32 + WMMA.
//
//   h   = f2fix(x @ w1 + b1); h = f2fix(pwl_gelu(h)); h = f2fix(h)
//   out = f2fix(h @ w2 + b2)
//
// Shapes: x [16384,1024], w1 [1024,4096], w2 [4096,1024].  ~275 GFLOP.
// Strategy: fp32 GEMM emulated with fp16 split-precision WMMA
// (v_wmma_f32_16x16x32_f16). GEMM1: A,B both split hi/lo -> 3 WMMA/k-step.
// GEMM2: h is quantized to 1/256 and |h|<8 so it is EXACT in fp16 -> A single,
// B split -> 2 WMMA/k-step. Weights pre-transposed to [N][K] so every WMMA
// fragment is a contiguous per-lane load (no LDS needed; weights are
// L2-resident: ~32 MB split fp16 vs 192 MB L2).
// ---------------------------------------------------------------------------

typedef _Float16 v16h __attribute__((ext_vector_type(16)));
typedef _Float16 v8h  __attribute__((ext_vector_type(8)));
typedef float    v8f  __attribute__((ext_vector_type(8)));

static constexpr int DIM  = 1024;
static constexpr int FF   = 4096;
static constexpr int MTOK = 32 * 512;   // 16384 tokens

// gelu(x) = x * Phi(x) at x = -4 + 0.25*i, i = 0..32  (matches _PWL_Y)
__constant__ float c_pwl[33] = {
    -1.2668497e-04f, -3.3156483e-04f, -8.1420171e-04f, -1.8753311e-03f,
    -4.0496940e-03f, -8.1943480e-03f, -1.5524163e-02f, -2.7505066e-02f,
    -4.5500264e-02f, -7.0103530e-02f, -1.0021080e-01f, -1.3206222e-01f,
    -1.5865525e-01f, -1.6997052e-01f, -1.5426877e-01f, -1.0032342e-01f,
     0.0000000e+00f,  1.4967658e-01f,  3.4573123e-01f,  5.8002948e-01f,
     8.4134475e-01f,  1.1179378e+00f,  1.3997892e+00f,  1.6798965e+00f,
     1.9544997e+00f,  2.2224950e+00f,  2.4844759e+00f,  2.7418057e+00f,
     2.9959503e+00f,  3.2481247e+00f,  3.4991858e+00f,  3.7496684e+00f,
     3.9998733e+00f};

__device__ __forceinline__ float f2fix(float x) {
    return rintf(x * 256.0f) * 0.00390625f;   // round-to-nearest-even, step 2^-8
}

__device__ __forceinline__ v8f wmma_f16(v16h a, v16h b, v8f c) {
    return __builtin_amdgcn_wmma_f32_16x16x32_f16(false, a, false, b,
                                                  (short)0, c, false, false);
}

// A-matrix 16x32 fp16 fragment (row-major source, leading dim ldk):
// lane<16: row = l15, K in {k0..k0+15}; lane>=16: K base +16.
__device__ __forceinline__ v16h load_a_frag(const _Float16* __restrict__ p) {
    v8h p0 = *(const v8h*)(p);
    v8h p1 = *(const v8h*)(p + 8);
    return __builtin_shufflevector(p0, p1, 0, 1, 2, 3, 4, 5, 6, 7,
                                   8, 9, 10, 11, 12, 13, 14, 15);
}

// ---------------------------------------------------------------------------
// Prep: fp32 -> fp16 hi/lo split (elementwise)
// ---------------------------------------------------------------------------
__global__ __launch_bounds__(256) void split_f32(const float* __restrict__ in,
                                                 _Float16* __restrict__ oh,
                                                 _Float16* __restrict__ ol,
                                                 long n) {
    long i = (long)blockIdx.x * blockDim.x + threadIdx.x;
    if (i < n) {
        float v = in[i];
        _Float16 h = (_Float16)v;
        oh[i] = h;
        ol[i] = (_Float16)(v - (float)h);
    }
}

// ---------------------------------------------------------------------------
// Prep: fp32 [R][C] -> fp16 hi/lo transposed [C][R]  (LDS-tiled)
// ---------------------------------------------------------------------------
__global__ __launch_bounds__(256) void split_transpose(
    const float* __restrict__ in, _Float16* __restrict__ oh,
    _Float16* __restrict__ ol, int R, int C) {
    __shared__ float tile[32][33];
    const int c0 = blockIdx.x * 32, r0 = blockIdx.y * 32;
    const int tx = threadIdx.x, ty = threadIdx.y;
#pragma unroll
    for (int j = 0; j < 32; j += 8)
        tile[ty + j][tx] = in[(long)(r0 + ty + j) * C + c0 + tx];
    __syncthreads();
#pragma unroll
    for (int j = 0; j < 32; j += 8) {
        int c = c0 + ty + j;
        float v = tile[tx][ty + j];
        _Float16 h = (_Float16)v;
        oh[(long)c * R + r0 + tx] = h;
        ol[(long)c * R + r0 + tx] = (_Float16)(v - (float)h);
    }
}

// ---------------------------------------------------------------------------
// GEMM1 fused: h = f2fix(pwl_gelu(f2fix(x@w1 + b1)))  -> fp16 (exact)
// Block: 256 thr = 8 waves (2 M x 4 N). Wave tile 32x64. Block tile 64x256.
// ---------------------------------------------------------------------------
__global__ __launch_bounds__(256) void gemm1_fused(
    const _Float16* __restrict__ xh, const _Float16* __restrict__ xl,
    const _Float16* __restrict__ w1th, const _Float16* __restrict__ w1tl,
    const float* __restrict__ b1, _Float16* __restrict__ hbuf) {
    const int lane = threadIdx.x & 31;
    const int wave = threadIdx.x >> 5;
    const int wm = wave & 1, wn = wave >> 1;
    const int mbase = blockIdx.x * 64 + wm * 32;
    const int nbase = blockIdx.y * 256 + wn * 64;
    const int l15 = lane & 15, hi = lane >> 4;
    const int kA = hi * 16;   // A fragment K sub-offset
    const int kB = hi * 16;   // B fragment K sub-offset

    const long arow0 = (long)(mbase + l15) * DIM;
    const long arow1 = (long)(mbase + 16 + l15) * DIM;
    long brow[4];
#pragma unroll
    for (int j = 0; j < 4; ++j) brow[j] = (long)(nbase + 16 * j + l15) * DIM;

    v8f acc[2][4];
#pragma unroll
    for (int t = 0; t < 2; ++t)
#pragma unroll
        for (int j = 0; j < 4; ++j) acc[t][j] = (v8f){};

    for (int kt = 0; kt < DIM; kt += 32) {
        v16h a0h = load_a_frag(xh + arow0 + kt + kA);
        v16h a0l = load_a_frag(xl + arow0 + kt + kA);
        v16h a1h = load_a_frag(xh + arow1 + kt + kA);
        v16h a1l = load_a_frag(xl + arow1 + kt + kA);
#pragma unroll
        for (int j = 0; j < 4; ++j) {
            v16h bh = *(const v16h*)(w1th + brow[j] + kt + kB);
            v16h bl = *(const v16h*)(w1tl + brow[j] + kt + kB);
            // fp32 emulation: hi*hi + lo*hi + hi*lo (lo*lo negligible)
            acc[0][j] = wmma_f16(a0h, bh, acc[0][j]);
            acc[0][j] = wmma_f16(a0l, bh, acc[0][j]);
            acc[0][j] = wmma_f16(a0h, bl, acc[0][j]);
            acc[1][j] = wmma_f16(a1h, bh, acc[1][j]);
            acc[1][j] = wmma_f16(a1l, bh, acc[1][j]);
            acc[1][j] = wmma_f16(a1h, bl, acc[1][j]);
        }
    }

    // Epilogue: +b1, quantize, PWL-GELU, quantize, store fp16 (exact).
    // C layout: VGPR r holds row (r + 8*(lane>=16)), col = lane&15.
#pragma unroll
    for (int t = 0; t < 2; ++t) {
        const int mrow = mbase + 16 * t + 8 * hi;
#pragma unroll
        for (int j = 0; j < 4; ++j) {
            const int n = nbase + 16 * j + l15;
            const float bias = b1[n];
#pragma unroll
            for (int r = 0; r < 8; ++r) {
                float q = f2fix(acc[t][j][r] + bias);
                float y;
                if (q > 4.0f) {
                    y = q;
                } else if (q < -4.0f) {
                    y = 0.0f;
                } else {
                    float tt = (q + 4.0f) * 4.0f;       // in [0,32]
                    int idx = (int)tt;
                    idx = idx > 31 ? 31 : idx;
                    float fr = tt - (float)idx;
                    float y0 = c_pwl[idx];
                    y = y0 + fr * (c_pwl[idx + 1] - y0);
                }
                y = f2fix(y);
                hbuf[(long)(mrow + r) * FF + n] = (_Float16)y;
            }
        }
    }
}

// ---------------------------------------------------------------------------
// GEMM2 fused: out = f2fix(h@w2 + b2)   (h exact fp16; w2 split hi/lo)
// ---------------------------------------------------------------------------
__global__ __launch_bounds__(256) void gemm2_fused(
    const _Float16* __restrict__ hbuf, const _Float16* __restrict__ w2th,
    const _Float16* __restrict__ w2tl, const float* __restrict__ b2,
    float* __restrict__ out) {
    const int lane = threadIdx.x & 31;
    const int wave = threadIdx.x >> 5;
    const int wm = wave & 1, wn = wave >> 1;
    const int mbase = blockIdx.x * 64 + wm * 32;
    const int nbase = blockIdx.y * 256 + wn * 64;
    const int l15 = lane & 15, hi = lane >> 4;
    const int kA = hi * 16;
    const int kB = hi * 16;

    const long arow0 = (long)(mbase + l15) * FF;
    const long arow1 = (long)(mbase + 16 + l15) * FF;
    long brow[4];
#pragma unroll
    for (int j = 0; j < 4; ++j) brow[j] = (long)(nbase + 16 * j + l15) * FF;

    v8f acc[2][4];
#pragma unroll
    for (int t = 0; t < 2; ++t)
#pragma unroll
        for (int j = 0; j < 4; ++j) acc[t][j] = (v8f){};

    for (int kt = 0; kt < FF; kt += 32) {
        v16h a0 = load_a_frag(hbuf + arow0 + kt + kA);
        v16h a1 = load_a_frag(hbuf + arow1 + kt + kA);
#pragma unroll
        for (int j = 0; j < 4; ++j) {
            v16h bh = *(const v16h*)(w2th + brow[j] + kt + kB);
            v16h bl = *(const v16h*)(w2tl + brow[j] + kt + kB);
            acc[0][j] = wmma_f16(a0, bh, acc[0][j]);
            acc[0][j] = wmma_f16(a0, bl, acc[0][j]);
            acc[1][j] = wmma_f16(a1, bh, acc[1][j]);
            acc[1][j] = wmma_f16(a1, bl, acc[1][j]);
        }
    }

#pragma unroll
    for (int t = 0; t < 2; ++t) {
        const int mrow = mbase + 16 * t + 8 * hi;
#pragma unroll
        for (int j = 0; j < 4; ++j) {
            const int n = nbase + 16 * j + l15;
            const float bias = b2[n];
#pragma unroll
            for (int r = 0; r < 8; ++r)
                out[(long)(mrow + r) * DIM + n] = f2fix(acc[t][j][r] + bias);
        }
    }
}

// ---------------------------------------------------------------------------
extern "C" void kernel_launch(void* const* d_in, const int* in_sizes, int n_in,
                              void* d_out, int out_size, void* d_ws,
                              size_t ws_size, hipStream_t stream) {
    const float* x  = (const float*)d_in[0];   // [16384,1024]
    const float* w1 = (const float*)d_in[1];   // [1024,4096]
    const float* b1 = (const float*)d_in[2];   // [4096]
    const float* w2 = (const float*)d_in[3];   // [4096,1024]
    const float* b2 = (const float*)d_in[4];   // [1024]
    float* out = (float*)d_out;                // [16384,1024] fp32

    // Workspace carve (total ~224 MB)
    char* ws = (char*)d_ws;
    size_t off = 0;
    auto carve = [&](size_t bytes) -> void* {
        void* p = ws + off;
        off = (off + bytes + 255) & ~(size_t)255;
        return p;
    };
    _Float16* xh   = (_Float16*)carve((size_t)MTOK * DIM * 2);  // 32 MB
    _Float16* xl   = (_Float16*)carve((size_t)MTOK * DIM * 2);  // 32 MB
    _Float16* w1th = (_Float16*)carve((size_t)FF * DIM * 2);    //  8 MB
    _Float16* w1tl = (_Float16*)carve((size_t)FF * DIM * 2);    //  8 MB
    _Float16* w2th = (_Float16*)carve((size_t)DIM * FF * 2);    //  8 MB
    _Float16* w2tl = (_Float16*)carve((size_t)DIM * FF * 2);    //  8 MB
    _Float16* hbuf = (_Float16*)carve((size_t)MTOK * FF * 2);   // 128 MB
    (void)ws_size; (void)in_sizes; (void)n_in; (void)out_size;

    // 1) split x into fp16 hi/lo
    {
        long n = (long)MTOK * DIM;
        split_f32<<<dim3((unsigned)((n + 255) / 256)), dim3(256), 0, stream>>>(
            x, xh, xl, n);
    }
    // 2) split+transpose weights to [N][K] fp16 hi/lo
    split_transpose<<<dim3(FF / 32, DIM / 32), dim3(32, 8), 0, stream>>>(
        w1, w1th, w1tl, DIM, FF);
    split_transpose<<<dim3(DIM / 32, FF / 32), dim3(32, 8), 0, stream>>>(
        w2, w2th, w2tl, FF, DIM);

    // 3) GEMM1 + quantize + PWL-GELU + quantize
    gemm1_fused<<<dim3(MTOK / 64, FF / 256), dim3(256), 0, stream>>>(
        xh, xl, w1th, w1tl, b1, hbuf);

    // 4) GEMM2 + quantize
    gemm2_fused<<<dim3(MTOK / 64, DIM / 256), dim3(256), 0, stream>>>(
        hbuf, w2th, w2tl, b2, out);
}
